// TransformerLayerShardV2_45131516347066
// MI455X (gfx1250) — matrix operand order repaired and checked by reading, not compiled, not run on hardware
//
#include <hip/hip_runtime.h>
#include <cstdint>
#include <cstddef>

// ---------------------------------------------------------------------------
// Types / WMMA helpers (CDNA5 / gfx1250, wave32)
// ---------------------------------------------------------------------------
typedef __attribute__((ext_vector_type(16))) _Float16 v16h;
typedef __attribute__((ext_vector_type(8)))  _Float16 v8h;
typedef __attribute__((ext_vector_type(8)))  float    v8f;
typedef __attribute__((ext_vector_type(4)))  int      v4i;

union V16U { v16h v; v8h h[2]; };

__device__ __forceinline__ v16h mk16(const _Float16* lo, const _Float16* hi) {
  V16U u;
  u.h[0] = *(const v8h*)lo;
  u.h[1] = *(const v8h*)hi;
  return u.v;
}

__device__ __forceinline__ v8f wmma16(v16h a, v16h b, v8f c) {
  // v_wmma_f32_16x16x32_f16  (8-arg form: neg_a, A, neg_b, B, c_mod, C, reuse_a, reuse_b)
  return __builtin_amdgcn_wmma_f32_16x16x32_f16(false, a, false, b, (short)0, c, false, false);
}

__device__ __forceinline__ v8f zero8() {
  v8f z = {0.f, 0.f, 0.f, 0.f, 0.f, 0.f, 0.f, 0.f};
  return z;
}

// ---------------------------------------------------------------------------
// Async global->LDS copies (CDNA5 GLOBAL_LOAD_ASYNC_TO_LDS_B128, ASYNCcnt).
// Guarded with __has_builtin so the file compiles on toolchains without them.
// Builtin signature (from hipcc diagnostic): param0 is int32x4* in AS(1).
// ---------------------------------------------------------------------------
#if defined(__has_builtin)
#if __has_builtin(__builtin_amdgcn_global_load_async_to_lds_b128) && \
    __has_builtin(__builtin_amdgcn_s_wait_asynccnt)
#define USE_ASYNC_LDS 1
#endif
#endif
#ifndef USE_ASYNC_LDS
#define USE_ASYNC_LDS 0
#endif

#if USE_ASYNC_LDS
typedef __attribute__((address_space(1))) v4i* g_v4i_p;
typedef __attribute__((address_space(3))) v4i* l_v4i_p;
__device__ __forceinline__ void async_cp16(const _Float16* g, _Float16* l) {
  __builtin_amdgcn_global_load_async_to_lds_b128((g_v4i_p)g, (l_v4i_p)l, 0, 0);
}
__device__ __forceinline__ void wait_async() {
  __builtin_amdgcn_s_wait_asynccnt(0);
}
#else
__device__ __forceinline__ void wait_async() {}
#endif

// ---------------------------------------------------------------------------
// Problem constants
// ---------------------------------------------------------------------------
#define BB    2
#define SS    2048
#define DM    1024
#define FIN   11264          // 16*64*3 + 1024*8
#define HH    8              // MP groups act as heads (d_head = 128)
#define DH    128
#define DCAT  5120           // 8 * (128 attn + 512 glu)
#define ROWS  4096           // B * S
#define GRP   1408           // FIN / 8
#define LNEPS 1e-5f

// ---------------------------------------------------------------------------
// K0: transpose + f32->f16 convert:  out[c*R + r] = (f16) in[r*C + c]
// ---------------------------------------------------------------------------
__global__ __launch_bounds__(256) void tls_transpose_cvt(
    const float* __restrict__ in, _Float16* __restrict__ out, int R, int C) {
  __shared__ float tile[32][33];
  const int c0 = blockIdx.x * 32, r0 = blockIdx.y * 32;
  const int tx = threadIdx.x & 31, ty = threadIdx.x >> 5; // 32 x 8
#pragma unroll
  for (int j = 0; j < 4; ++j)
    tile[ty + 8 * j][tx] = in[(size_t)(r0 + ty + 8 * j) * C + (c0 + tx)];
  __syncthreads();
#pragma unroll
  for (int j = 0; j < 4; ++j)
    out[(size_t)(c0 + ty + 8 * j) * R + (r0 + tx)] = (_Float16)tile[tx][ty + 8 * j];
}

// ---------------------------------------------------------------------------
// K1: LayerNorm (one block per row of 1024) -> f16
// ---------------------------------------------------------------------------
__global__ __launch_bounds__(256) void tls_layernorm(
    const float* __restrict__ x, const float* __restrict__ sc,
    const float* __restrict__ of, _Float16* __restrict__ h) {
  const int row = blockIdx.x;
  const int tid = threadIdx.x;
  const float4 v = ((const float4*)(x + (size_t)row * DM))[tid];
  float s  = v.x + v.y + v.z + v.w;
  float s2 = v.x * v.x + v.y * v.y + v.z * v.z + v.w * v.w;
  __shared__ float rs[256], rq[256];
  rs[tid] = s; rq[tid] = s2;
  __syncthreads();
  for (int o = 128; o > 0; o >>= 1) {
    if (tid < o) { rs[tid] += rs[tid + o]; rq[tid] += rq[tid + o]; }
    __syncthreads();
  }
  const float mu  = rs[0] * (1.f / DM);
  const float var = rq[0] * (1.f / DM) - mu * mu;
  const float r   = rsqrtf(var + LNEPS);
  const int c = tid * 4;
  const float vv[4] = {v.x, v.y, v.z, v.w};
#pragma unroll
  for (int j = 0; j < 4; ++j)
    h[(size_t)row * DM + c + j] = (_Float16)((vv[j] - mu) * r * sc[c + j] + of[c + j]);
}

// ---------------------------------------------------------------------------
// K2/K5: WMMA GEMM  C[M,N] = A[M,K] * B[K,N] + bias,  B given transposed Bt[N,K].
// Block tile 128x128, BK=64, double-buffered LDS, 8 waves (2M x 4N),
// wave tile 64x32 -> 8 f32 accumulators (4 M-tiles x 2 N-tiles).
// ---------------------------------------------------------------------------
template <int OUTF16>
__global__ __launch_bounds__(256) void tls_gemm_bt(
    const _Float16* __restrict__ A, const _Float16* __restrict__ Bt,
    const float* __restrict__ bias, _Float16* __restrict__ Ch,
    float* __restrict__ Cf, int M, int N, int K) {
  __shared__ alignas(16) _Float16 As[2][128][72];   // pad 72: conflict-free frags
  __shared__ alignas(16) _Float16 Bs[2][128][72];

  const int tid  = threadIdx.x;
  const int w    = tid >> 5;
  const int lane = tid & 31;
  const int half = lane >> 4;
  const int ln   = lane & 15;
  const int wm   = w & 1;          // 0..1  (M)
  const int wn   = w >> 1;         // 0..3  (N)
  const int m0   = blockIdx.y * 128;
  const int n0   = blockIdx.x * 128;
  const int KT   = K >> 6;

  v8f acc[4][2];
#pragma unroll
  for (int i = 0; i < 4; ++i)
#pragma unroll
    for (int j = 0; j < 2; ++j) acc[i][j] = zero8();

  const int srow  = tid >> 1;      // 0..127
  const int spart = tid & 1;       // 0..1 -> 32 halfs each

  auto stage = [&](int buf, int kt) {
    const _Float16* pa = A + (size_t)(m0 + srow) * K + kt * 64 + spart * 32;
    const _Float16* pb = Bt + (size_t)(n0 + srow) * K + kt * 64 + spart * 32;
#if USE_ASYNC_LDS
#pragma unroll
    for (int j = 0; j < 4; ++j) {
      async_cp16(pa + j * 8, &As[buf][srow][spart * 32 + j * 8]);
      async_cp16(pb + j * 8, &Bs[buf][srow][spart * 32 + j * 8]);
    }
#else
    const v8h* va = (const v8h*)pa;
    const v8h* vb = (const v8h*)pb;
#pragma unroll
    for (int j = 0; j < 4; ++j) {
      *(v8h*)&As[buf][srow][spart * 32 + j * 8] = va[j];
      *(v8h*)&Bs[buf][srow][spart * 32 + j * 8] = vb[j];
    }
#endif
  };

  stage(0, 0);
  for (int kt = 0; kt < KT; ++kt) {
    wait_async();                 // drain this thread's async copies, then rendezvous
    __syncthreads();
    const int cur = kt & 1;
    if (kt + 1 < KT) stage(cur ^ 1, kt + 1);
    if (kt + 2 < KT) {  // global_prefetch_b8 of the tile after next
      __builtin_prefetch(A + (size_t)(m0 + srow) * K + (kt + 2) * 64 + spart * 32, 0, 1);
      __builtin_prefetch(Bt + (size_t)(n0 + srow) * K + (kt + 2) * 64 + spart * 32, 0, 1);
    }
#pragma unroll
    for (int kk = 0; kk < 2; ++kk) {
      v16h aF[4], bF[2];
#pragma unroll
      for (int mt = 0; mt < 4; ++mt) {
        const _Float16* p = &As[cur][wm * 64 + mt * 16 + ln][kk * 32];
        aF[mt] = mk16(p + half * 8, p + 16 + half * 8);
      }
#pragma unroll
      for (int nt = 0; nt < 2; ++nt) {
        const _Float16* p = &Bs[cur][wn * 32 + nt * 16 + ln][kk * 32];
        bF[nt] = mk16(p + half * 8, p + 16 + half * 8);
      }
#pragma unroll
      for (int mt = 0; mt < 4; ++mt)
#pragma unroll
        for (int nt = 0; nt < 2; ++nt)
          acc[mt][nt] = wmma16(aF[mt], bF[nt], acc[mt][nt]);
    }
  }

#pragma unroll
  for (int nt = 0; nt < 2; ++nt) {
    const int col = n0 + wn * 32 + nt * 16 + ln;
    const float bv = bias[col];
#pragma unroll
    for (int mt = 0; mt < 4; ++mt) {
#pragma unroll
      for (int i = 0; i < 8; ++i) {
        const int row = m0 + wm * 64 + mt * 16 + i + 8 * half; // D layout
        const float v = acc[mt][nt][i] + bv;
        if (OUTF16) Ch[(size_t)row * N + col] = (_Float16)v;
        else        Cf[(size_t)row * N + col] = v;
      }
    }
  }
}

// ---------------------------------------------------------------------------
// K3: rearrange proj -> Qr/Kr (rotary, [b,g,t,128]), Vt ([b,g,128,t]),
//     and GLU part of the concat matrix Cc[4096, 5120].
// ---------------------------------------------------------------------------
#define NQ_ELEM (4194304ull)               // 2*8*2048*128
#define NG_ELEM (16777216ull)              // 4096*4096
#define REARR_TOTAL (3ull * NQ_ELEM + NG_ELEM)

__global__ __launch_bounds__(256) void tls_rearrange(
    const _Float16* __restrict__ proj, _Float16* __restrict__ Qr,
    _Float16* __restrict__ Kr, _Float16* __restrict__ VtG,
    _Float16* __restrict__ Cc) {
  const size_t idx = (size_t)blockIdx.x * 256 + threadIdx.x;
  if (idx >= REARR_TOTAL) return;

  if (idx < 2 * NQ_ELEM) {                  // Q or K with GPT-J rotary (first 64 dims)
    const bool isK = idx >= NQ_ELEM;
    const size_t j = isK ? idx - NQ_ELEM : idx;   // ((b*8+g)*2048 + t)*128 + d
    const int d  = (int)(j & 127);
    const size_t r = j >> 7;
    const int t  = (int)(r & 2047);
    const int bg = (int)(r >> 11);
    const int g  = bg & 7, b = bg >> 3;
    const size_t prow = (size_t)(b * SS + t) * FIN + (size_t)g * GRP + (isK ? 256 : 0);
    float outv;
    if (d < 64) {
      const int i = d >> 1, base = d & ~1;
      const float x1 = (float)proj[prow + base];
      const float x2 = (float)proj[prow + base + 1];
      const float invf = __expf(-(float)i * 0.28782313662425575f); // ln(1e4)/32
      const float th = (float)t * invf;
      const float sn = __sinf(th), cs = __cosf(th);
      outv = (d & 1) ? (x2 * cs + x1 * sn) : (x1 * cs - x2 * sn);
    } else {
      outv = (float)proj[prow + d];
    }
    (isK ? Kr : Qr)[j] = (_Float16)outv;
  } else if (idx < 3 * NQ_ELEM) {           // V transposed (d-major): [(b*8+g)*128+d]*2048 + t
    const size_t j = idx - 2 * NQ_ELEM;
    const int t  = (int)(j & 2047);
    const size_t r = j >> 11;
    const int d  = (int)(r & 127);
    const int bg = (int)(r >> 7);
    const int g  = bg & 7, b = bg >> 3;
    VtG[j] = proj[(size_t)(b * SS + t) * FIN + (size_t)g * GRP + 128 + d];
  } else {                                  // GLU -> concat columns g*640 + 128 + q
    const size_t j = idx - 3 * NQ_ELEM;
    const int col = (int)(j & 4095);
    const size_t row = j >> 12;
    const int g = col >> 9, q = col & 511;
    const size_t pb = row * FIN + (size_t)g * GRP + 384;
    const float fo = (float)proj[pb + q];
    const float x  = (float)proj[pb + 512 + q];
    const float gel = 0.5f * x * (1.f + tanhf(0.7978845608f * (x + 0.044715f * x * x * x)));
    Cc[row * DCAT + g * 640 + 128 + q] = (_Float16)(fo * gel);
  }
}

// ---------------------------------------------------------------------------
// K4: causal flash attention per (query-block 128, head, batch).
// 8 waves x 16-query tiles; 32-key chunks via LDS; logits = 4 chained WMMAs,
// online softmax, P re-shaped through wave-private LDS into A-layout, PV = 8 WMMAs.
// Interior chunks take a maskless fast path (wave-uniform branch).
// ---------------------------------------------------------------------------
__global__ __launch_bounds__(256) void tls_attention(
    const _Float16* __restrict__ Qr, const _Float16* __restrict__ Kr,
    const _Float16* __restrict__ VtG, const float* __restrict__ ab,
    _Float16* __restrict__ Cc) {
  __shared__ alignas(16) _Float16 Kc[32][136];    // 32 keys x 128d, padded
  __shared__ alignas(16) _Float16 Vl[128][40];    // 128 d   x 32 keys, padded
  __shared__ alignas(16) _Float16 Ps[8][16][40];  // per-wave P staging

  const int w    = threadIdx.x >> 5;
  const int lane = threadIdx.x & 31;
  const int half = lane >> 4;
  const int ln   = lane & 15;
  const int q0b  = blockIdx.x * 128;
  const int g    = blockIdx.y;
  const int b    = blockIdx.z;
  const int q0w  = q0b + w * 16;

  const size_t headoff = (size_t)(b * HH + g) * SS * DH;
  const _Float16* Qh = Qr + headoff;
  const _Float16* Kh = Kr + headoff;
  const _Float16* Vh = VtG + headoff;

  // Q fragments (A-layout), 16 queries x 128 d = 4 K-chunks
  v16h aQ[4];
  {
    const _Float16* qrow = Qh + (size_t)(q0w + ln) * DH;
#pragma unroll
    for (int c = 0; c < 4; ++c)
      aQ[c] = mk16(qrow + c * 32 + half * 8, qrow + c * 32 + 16 + half * 8);
  }

  v8f Ov[8];
  float m_[8], l_[8];
#pragma unroll
  for (int i = 0; i < 8; ++i) { Ov[i] = zero8(); m_[i] = -1e30f; l_[i] = 0.f; }

  const int nkc = q0b / 32 + 4;           // causal: keys 0 .. q0b+127
  for (int kc = 0; kc < nkc; ++kc) {
    __syncthreads();                      // previous compute done before overwrite
    {   // cooperative staging
      const int tt = threadIdx.x;
      const int kr = tt >> 3, part = tt & 7;
      const _Float16* ks = Kh + (size_t)(kc * 32 + kr) * DH + part * 16;
      const int dr = tt >> 1, p2 = tt & 1;
      const _Float16* vsrc = Vh + (size_t)dr * SS + kc * 32 + p2 * 16;
#if USE_ASYNC_LDS
      async_cp16(ks,     &Kc[kr][part * 16]);
      async_cp16(ks + 8, &Kc[kr][part * 16 + 8]);
      async_cp16(vsrc,     &Vl[dr][p2 * 16]);
      async_cp16(vsrc + 8, &Vl[dr][p2 * 16 + 8]);
#else
      const v8h* src = (const v8h*)ks;
      *(v8h*)&Kc[kr][part * 16]     = src[0];
      *(v8h*)&Kc[kr][part * 16 + 8] = src[1];
      const v8h* vs = (const v8h*)vsrc;
      *(v8h*)&Vl[dr][p2 * 16]     = vs[0];
      *(v8h*)&Vl[dr][p2 * 16 + 8] = vs[1];
#endif
    }
    wait_async();
    __syncthreads();
    if (kc * 32 > q0w + 15) continue;     // nothing unmasked for this wave

    // logits: two 16-key tiles, K-dim = 128 (4 WMMA steps)
    v8f s0 = zero8(), s1 = zero8();
#pragma unroll
    for (int ds = 0; ds < 4; ++ds) {
      const _Float16* p0 = &Kc[ln][ds * 32];
      const _Float16* p1 = &Kc[16 + ln][ds * 32];
      s0 = wmma16(aQ[ds], mk16(p0 + half * 8, p0 + 16 + half * 8), s0);
      s1 = wmma16(aQ[ds], mk16(p1 + half * 8, p1 + 16 + half * 8), s1);
    }

    const int k0p = kc * 32 + ln;
    const bool needMask = (kc * 32 + 31) > q0w;   // wave-uniform: diagonal chunk?
    float alpha[8];
#pragma unroll
    for (int i = 0; i < 8; ++i) {
      const int q = q0w + i + 8 * half;   // D-layout row
      const float b0 = ab[(size_t)q * SS + k0p];
      const float b1 = ab[(size_t)q * SS + k0p + 16];
      float v0 = s0[i] * 0.125f + b0;
      float v1 = s1[i] * 0.125f + b1;
      if (needMask) {                     // only one diagonal chunk per wave
        v0 += (k0p      > q) ? -1e10f : 0.f;
        v1 += (k0p + 16 > q) ? -1e10f : 0.f;
      }
      float rm = fmaxf(v0, v1);
#pragma unroll
      for (int o = 8; o > 0; o >>= 1) rm = fmaxf(rm, __shfl_xor(rm, o, 16));
      const float mn = fmaxf(m_[i], rm);
      alpha[i] = __expf(m_[i] - mn);
      const float p0 = __expf(v0 - mn), p1 = __expf(v1 - mn);
      float rs = p0 + p1;
#pragma unroll
      for (int o = 8; o > 0; o >>= 1) rs += __shfl_xor(rs, o, 16);
      l_[i] = l_[i] * alpha[i] + rs;
      m_[i] = mn;
      Ps[w][i + 8 * half][ln]      = (_Float16)p0;
      Ps[w][i + 8 * half][16 + ln] = (_Float16)p1;
    }
    __builtin_amdgcn_wave_barrier();      // keep LDS store->load ordered (same wave, DS in-order)

    const v16h aP = mk16(&Ps[w][ln][half * 8], &Ps[w][ln][16 + half * 8]);
#pragma unroll
    for (int dt = 0; dt < 8; ++dt) {
#pragma unroll
      for (int i = 0; i < 8; ++i) Ov[dt][i] *= alpha[i];
      const _Float16* pv = &Vl[dt * 16 + ln][0];
      Ov[dt] = wmma16(aP, mk16(pv + half * 8, pv + 16 + half * 8), Ov[dt]);
    }
  }

  float inv[8];
#pragma unroll
  for (int i = 0; i < 8; ++i) inv[i] = 1.f / l_[i];
#pragma unroll
  for (int dt = 0; dt < 8; ++dt)
#pragma unroll
    for (int i = 0; i < 8; ++i) {
      const int row = b * SS + q0w + i + 8 * half;
      Cc[(size_t)row * DCAT + g * 640 + dt * 16 + ln] = (_Float16)(Ov[dt][i] * inv[i]);
    }
}

// ---------------------------------------------------------------------------
// Host launcher
// ---------------------------------------------------------------------------
extern "C" void kernel_launch(void* const* d_in, const int* in_sizes, int n_in,
                              void* d_out, int out_size, void* d_ws, size_t ws_size,
                              hipStream_t stream) {
  (void)in_sizes; (void)n_in; (void)out_size; (void)ws_size;
  const float* x     = (const float*)d_in[0];
  const float* ab    = (const float*)d_in[1];
  const float* ln_s  = (const float*)d_in[2];
  const float* ln_o  = (const float*)d_in[3];
  const float* w_in  = (const float*)d_in[4];
  const float* b_in  = (const float*)d_in[5];
  const float* w_out = (const float*)d_in[6];
  const float* b_out = (const float*)d_in[7];
  float* out = (float*)d_out;

  char* ws = (char*)d_ws;
  size_t off = 0;
  auto carve = [&](size_t bytes) -> void* {
    void* p = ws + off;
    off += (bytes + 255) & ~(size_t)255;
    return p;
  };
  _Float16* hA     = (_Float16*)carve((size_t)ROWS * DM * 2);        //  8 MB
  _Float16* w_inT  = (_Float16*)carve((size_t)FIN * DM * 2);         // 23 MB (Bt for GEMM1)
  _Float16* w_outT = (_Float16*)carve((size_t)DM * DCAT * 2);        // 10.5 MB (Bt for GEMM2)
  _Float16* proj   = (_Float16*)carve((size_t)ROWS * FIN * 2);       // 92 MB
  _Float16* Qr     = (_Float16*)carve(NQ_ELEM * 2);                  //  8 MB
  _Float16* Kr     = (_Float16*)carve(NQ_ELEM * 2);                  //  8 MB
  _Float16* VtG    = (_Float16*)carve(NQ_ELEM * 2);                  //  8 MB
  _Float16* Cc     = (_Float16*)carve((size_t)ROWS * DCAT * 2);      // 42 MB

  // K0: weight transpose + convert
  tls_transpose_cvt<<<dim3(FIN / 32, DM / 32), 256, 0, stream>>>(w_in, w_inT, DM, FIN);
  tls_transpose_cvt<<<dim3(DM / 32, DCAT / 32), 256, 0, stream>>>(w_out, w_outT, DCAT, DM);

  // K1: layernorm -> f16
  tls_layernorm<<<ROWS, 256, 0, stream>>>(x, ln_s, ln_o, hA);

  // K2: proj = h @ w_in + b_in  (f16 out)
  tls_gemm_bt<1><<<dim3(FIN / 128, ROWS / 128), 256, 0, stream>>>(
      hA, w_inT, b_in, proj, nullptr, ROWS, FIN, DM);

  // K3: rotary / V-transpose / GLU
  tls_rearrange<<<(unsigned)(REARR_TOTAL / 256), 256, 0, stream>>>(proj, Qr, Kr, VtG, Cc);

  // K4: causal flash attention -> attn part of Cc
  tls_attention<<<dim3(SS / 128, HH, BB), 256, 0, stream>>>(Qr, Kr, VtG, ab, Cc);

  // K5: out = Cc @ w_out + b_out (f32 out)
  tls_gemm_bt<0><<<dim3(DM / 128, ROWS / 128), 256, 0, stream>>>(
      Cc, w_outT, b_out, nullptr, out, ROWS, DM, DCAT);
}
